// SparseAttention_49289044689501
// MI455X (gfx1250) — matrix operand order repaired and checked by reading, not compiled, not run on hardware
//
#include <hip/hip_runtime.h>
#include <math.h>

// ---------------------------------------------------------------------------
// Types for CDNA5 WMMA (wave32): bf16 A/B fragments (16 x bf16 per lane),
// f32 C/D accumulator (8 x f32 per lane).
// ---------------------------------------------------------------------------
typedef __attribute__((ext_vector_type(16))) __bf16 v16bf;
typedef __attribute__((ext_vector_type(8)))  __bf16 v8bf;
typedef __attribute__((ext_vector_type(8)))  float  v8f;

#define WMMA_BF16(a, b, c) \
  __builtin_amdgcn_wmma_f32_16x16x32_bf16(false, (a), false, (b), (short)0, (c), false, false)

// 8 WMMAs of a 32(M)x64(N) wave tile against one 32-deep K step.
#define GEMM8(acc, A0, A1, B0, B1, B2, B3)      \
  do {                                          \
    acc[0] = WMMA_BF16(A0, B0, acc[0]);         \
    acc[1] = WMMA_BF16(A0, B1, acc[1]);         \
    acc[2] = WMMA_BF16(A0, B2, acc[2]);         \
    acc[3] = WMMA_BF16(A0, B3, acc[3]);         \
    acc[4] = WMMA_BF16(A1, B0, acc[4]);         \
    acc[5] = WMMA_BF16(A1, B1, acc[5]);         \
    acc[6] = WMMA_BF16(A1, B2, acc[6]);         \
    acc[7] = WMMA_BF16(A1, B3, acc[7]);         \
  } while (0)

// A-matrix 16x32 bf16 fragment (ISA 7.12.2): lane L holds row M = L%16.
// element e -> K = (e>>3)*16 + (L>>4)*8 + (e&7)  => two contiguous 16B loads.
__device__ __forceinline__ v16bf load_a_frag(const __bf16* __restrict__ base, int ld) {
  const int lane = threadIdx.x & 31;
  const int r = lane & 15;
  const int half = lane >> 4;
  const __bf16* p = base + (size_t)r * ld + half * 8;
  v8bf lo = *(const v8bf*)(p);        // K = half*8 + 0..7
  v8bf hi = *(const v8bf*)(p + 16);   // K = 16 + half*8 + 0..7
  v16bf out;
#pragma unroll
  for (int i = 0; i < 8; ++i) { out[i] = lo[i]; out[8 + i] = hi[i]; }
  return out;
}

// B-matrix 32x16 bf16 fragment: lane L holds column N = L%16;
// element e -> K = (e>>3)*16 + (L>>4)*8 + (e&7) (two contiguous 16B loads).
__device__ __forceinline__ v16bf load_b_frag(const __bf16* __restrict__ base, int ld) {
  const int lane = threadIdx.x & 31;
  const int n = lane & 15;
  const int half = lane >> 4;
  const __bf16* p = base + (size_t)n * ld + half * 8;
  v8bf lo = *(const v8bf*)(p);        // K = half*8 + 0..7
  v8bf hi = *(const v8bf*)(p + 16);   // K = 16 + half*8 + 0..7
  v16bf out;
#pragma unroll
  for (int i = 0; i < 8; ++i) { out[i] = lo[i]; out[8 + i] = hi[i]; }
  return out;
}

// ---------------------------------------------------------------------------
// Kernel 0: f32 -> bf16 conversion
// ---------------------------------------------------------------------------
__global__ __launch_bounds__(256) void cvt_bf16(const float* __restrict__ in,
                                                __bf16* __restrict__ out, int n) {
  int i = blockIdx.x * 256 + threadIdx.x;
  if (i < n) out[i] = (__bf16)in[i];
}

// ---------------------------------------------------------------------------
// Kernel 1: QKV projection GEMM.  qkv[t,e] = sum_d x[t,d] * W_qkv[e,d]
// M=4096 tokens, N=3072, K=1024.  256 thr (8 waves); each wave owns a
// 32(M)x64(N) tile.  Ping-pong K pipeline: two fragment sets, each refilled
// by loads only (no reg-to-reg rotation -> no RA copies / hazard nops).
// ---------------------------------------------------------------------------
__global__ __launch_bounds__(256) void qkv_gemm(const __bf16* __restrict__ xb,
                                                const __bf16* __restrict__ wb,
                                                __bf16* __restrict__ qbuf,
                                                __bf16* __restrict__ kbuf,
                                                __bf16* __restrict__ vT) {
  const int col0 = blockIdx.x * 64;
  const int wave = threadIdx.x >> 5;
  const int row0 = blockIdx.y * 256 + wave * 32;
  const __bf16* Abase = xb + (size_t)row0 * 1024;
  const __bf16* Bbase = wb + (size_t)col0 * 1024;

  v8f acc[8] = {};
  // set A @ k=0, set B @ k=32
  v16bf a0A = load_a_frag(Abase, 1024);
  v16bf a1A = load_a_frag(Abase + (size_t)16 * 1024, 1024);
  v16bf b0A = load_b_frag(Bbase, 1024);
  v16bf b1A = load_b_frag(Bbase + (size_t)16 * 1024, 1024);
  v16bf b2A = load_b_frag(Bbase + (size_t)32 * 1024, 1024);
  v16bf b3A = load_b_frag(Bbase + (size_t)48 * 1024, 1024);
  v16bf a0B = load_a_frag(Abase + 32, 1024);
  v16bf a1B = load_a_frag(Abase + (size_t)16 * 1024 + 32, 1024);
  v16bf b0B = load_b_frag(Bbase + 32, 1024);
  v16bf b1B = load_b_frag(Bbase + (size_t)16 * 1024 + 32, 1024);
  v16bf b2B = load_b_frag(Bbase + (size_t)32 * 1024 + 32, 1024);
  v16bf b3B = load_b_frag(Bbase + (size_t)48 * 1024 + 32, 1024);

  for (int k0 = 64; k0 < 1024; k0 += 64) {
    __builtin_prefetch(Bbase + k0 + 64, 0, 1);   // global_prefetch_b8
    // consume set A, refill it from k0
    GEMM8(acc, a0A, a1A, b0A, b1A, b2A, b3A);
    a0A = load_a_frag(Abase + k0, 1024);
    a1A = load_a_frag(Abase + (size_t)16 * 1024 + k0, 1024);
    b0A = load_b_frag(Bbase + k0, 1024);
    b1A = load_b_frag(Bbase + (size_t)16 * 1024 + k0, 1024);
    b2A = load_b_frag(Bbase + (size_t)32 * 1024 + k0, 1024);
    b3A = load_b_frag(Bbase + (size_t)48 * 1024 + k0, 1024);
    // consume set B, refill it from k0+32
    GEMM8(acc, a0B, a1B, b0B, b1B, b2B, b3B);
    a0B = load_a_frag(Abase + k0 + 32, 1024);
    a1B = load_a_frag(Abase + (size_t)16 * 1024 + k0 + 32, 1024);
    b0B = load_b_frag(Bbase + k0 + 32, 1024);
    b1B = load_b_frag(Bbase + (size_t)16 * 1024 + k0 + 32, 1024);
    b2B = load_b_frag(Bbase + (size_t)32 * 1024 + k0 + 32, 1024);
    b3B = load_b_frag(Bbase + (size_t)48 * 1024 + k0 + 32, 1024);
  }
  // tail: k=960 (set A) and k=992 (set B)
  GEMM8(acc, a0A, a1A, b0A, b1A, b2A, b3A);
  GEMM8(acc, a0B, a1B, b0B, b1B, b2B, b3B);

  const int lane = threadIdx.x & 31;
  const int cb = lane & 15;
  const int rh = (lane >> 4) * 8;
#pragma unroll
  for (int mt = 0; mt < 2; ++mt) {
#pragma unroll
    for (int nt = 0; nt < 4; ++nt) {
#pragma unroll
      for (int r = 0; r < 8; ++r) {
        const int t = row0 + mt * 16 + rh + r;   // global token
        const int e = col0 + nt * 16 + cb;       // qkv channel (0..3071)
        const float val = acc[mt * 4 + nt][r];
        const int bb = t >> 11, n = t & 2047;
        const int which = e >> 10;
        const int hh = (e & 1023) >> 6;
        const int dd = e & 63;
        const int bh = bb * 16 + hh;
        if (which == 0)      qbuf[((size_t)bh * 2048 + n) * 64 + dd] = (__bf16)val;
        else if (which == 1) kbuf[((size_t)bh * 2048 + n) * 64 + dd] = (__bf16)val;
        else                 vT[((size_t)bh * 64 + dd) * 2048 + n] = (__bf16)fmaxf(val, 0.0f);
      }
    }
  }
}

// ---------------------------------------------------------------------------
// Kernel 2: sparse attention for one (b,h) and 16 query rows per workgroup.
// Phase A: scores s = (q.k^T)/32 via WMMA -> 16x2048 f32 in LDS (ping-pong).
// Phase B: per-row exact top-k threshold (k=204) via 32-bit radix select.
// Phase C: p = exp(s-m) masked, attn.relu(v) via WMMA (ping-pong), LDS f32
//          atomic cross-wave reduction, normalize, bf16 head output.
// ---------------------------------------------------------------------------
__global__ __launch_bounds__(256) void attn_kernel(const __bf16* __restrict__ qbuf,
                                                   const __bf16* __restrict__ kbuf,
                                                   const __bf16* __restrict__ vT,
                                                   __bf16* __restrict__ headout) {
  extern __shared__ char smem_raw[];
  float* sc     = (float*)smem_raw;                        // 16 x 2048 f32 (128 KB)
  float* rstat  = (float*)(smem_raw + 16 * 2048 * 4);      // [0..15]=thresh [16..31]=max [32..47]=1/sum
  float* outacc = rstat + 48;                              // 16 x 64 f32

  const int wg = blockIdx.x;         // 4096 = (b*h) * 128 i-tiles
  const int itile = wg & 127;
  const int bh = wg >> 7;
  const int i0 = itile * 16;
  const int wave = threadIdx.x >> 5;
  const int lane = threadIdx.x & 31;
  const int cb = lane & 15;
  const int rh = (lane >> 4) * 8;
  const int half = lane >> 4;

  const __bf16* qT = qbuf + ((size_t)bh * 2048 + i0) * 64;
  const __bf16* kb = kbuf + (size_t)bh * 2048 * 64;
  const __bf16* vb = vT + (size_t)bh * 64 * 2048;

  // ---- Phase A: scores, ping-pong over pairs of j-tiles ----------------
  v16bf a0 = load_a_frag(qT, 64);        // d = 0..31
  v16bf a1 = load_a_frag(qT + 32, 64);   // d = 32..63
  v16bf k0A = load_b_frag(kb + (size_t)wave * 16 * 64, 64);
  v16bf k1A = load_b_frag(kb + (size_t)wave * 16 * 64 + 32, 64);
  v16bf k0B = load_b_frag(kb + (size_t)(wave + 8) * 16 * 64, 64);
  v16bf k1B = load_b_frag(kb + (size_t)(wave + 8) * 16 * 64 + 32, 64);
  for (int jt = wave; jt < 128; jt += 16) {
    {
      v8f c = {};
      c = WMMA_BF16(a0, k0A, c);
      c = WMMA_BF16(a1, k1A, c);
      const int col = jt * 16 + cb;
#pragma unroll
      for (int r = 0; r < 8; ++r) sc[(rh + r) * 2048 + col] = c[r] * 0.03125f;
    }
    if (jt + 16 < 128) {
      k0A = load_b_frag(kb + (size_t)(jt + 16) * 16 * 64, 64);
      k1A = load_b_frag(kb + (size_t)(jt + 16) * 16 * 64 + 32, 64);
    }
    {
      v8f c = {};
      c = WMMA_BF16(a0, k0B, c);
      c = WMMA_BF16(a1, k1B, c);
      const int col = (jt + 8) * 16 + cb;
#pragma unroll
      for (int r = 0; r < 8; ++r) sc[(rh + r) * 2048 + col] = c[r] * 0.03125f;
    }
    if (jt + 24 < 128) {
      k0B = load_b_frag(kb + (size_t)(jt + 24) * 16 * 64, 64);
      k1B = load_b_frag(kb + (size_t)(jt + 24) * 16 * 64 + 32, 64);
    }
  }
  __syncthreads();

  // ---- Phase B: per-row threshold / max / exp-sum (2 rows per wave) ----
  for (int rr = 0; rr < 2; ++rr) {
    const int row = wave * 2 + rr;
    const float* srow = sc + row * 2048;
    unsigned key[64];
    float m = -3.4e38f;
#pragma unroll
    for (int t = 0; t < 64; ++t) {
      float s = srow[lane + t * 32];
      m = fmaxf(m, s);
      unsigned b = __float_as_uint(s);
      key[t] = (b & 0x80000000u) ? ~b : (b | 0x80000000u); // order-preserving key
    }
#pragma unroll
    for (int off = 16; off > 0; off >>= 1) m = fmaxf(m, __shfl_xor(m, off, 32));

    // largest T with count(key >= T) >= 204  ==  204-th largest key
    unsigned prefix = 0u;
    for (int bit = 31; bit >= 0; --bit) {
      const unsigned cand = prefix | (1u << bit);
      int cnt = 0;
#pragma unroll
      for (int t = 0; t < 64; ++t) cnt += (key[t] >= cand) ? 1 : 0;
#pragma unroll
      for (int off = 16; off > 0; off >>= 1) cnt += __shfl_xor(cnt, off, 32);
      if (cnt >= 204) prefix = cand;
    }
    const float thresh =
        __uint_as_float((prefix & 0x80000000u) ? (prefix & 0x7fffffffu) : ~prefix);

    float sum = 0.0f;
#pragma unroll
    for (int t = 0; t < 64; ++t) {
      float s = srow[lane + t * 32];
      if (s >= thresh) sum += __expf(s - m);
    }
#pragma unroll
    for (int off = 16; off > 0; off >>= 1) sum += __shfl_xor(sum, off, 32);
    if (lane == 0) { rstat[row] = thresh; rstat[16 + row] = m; rstat[32 + row] = 1.0f / sum; }
  }
  for (int idx = threadIdx.x; idx < 1024; idx += 256) outacc[idx] = 0.0f;
  __syncthreads();

  // ---- Phase C: out = softmax(masked) . relu(v), ping-pong -------------
  const float th_i = rstat[cb];       // this lane's A-row = cb
  const float m_i  = rstat[16 + cb];
  v8f acc2[4] = {};
  const int j0w = wave * 32;
  v16bf vA0 = load_b_frag(vb + (size_t)0 * 2048 + j0w, 2048);
  v16bf vA1 = load_b_frag(vb + (size_t)16 * 2048 + j0w, 2048);
  v16bf vA2 = load_b_frag(vb + (size_t)32 * 2048 + j0w, 2048);
  v16bf vA3 = load_b_frag(vb + (size_t)48 * 2048 + j0w, 2048);
  v16bf vB0 = load_b_frag(vb + (size_t)0 * 2048 + j0w + 256, 2048);
  v16bf vB1 = load_b_frag(vb + (size_t)16 * 2048 + j0w + 256, 2048);
  v16bf vB2 = load_b_frag(vb + (size_t)32 * 2048 + j0w + 256, 2048);
  v16bf vB3 = load_b_frag(vb + (size_t)48 * 2048 + j0w + 256, 2048);
  for (int jt = wave; jt < 64; jt += 16) {
    const int j0 = jt * 32;
    {
      v16bf pa;
#pragma unroll
      for (int g = 0; g < 2; ++g) {
#pragma unroll
        for (int w = 0; w < 8; ++w) {
          const int j = j0 + g * 16 + half * 8 + w;    // A-frag K mapping
          const float s = sc[cb * 2048 + j];
          const float p = (s >= th_i) ? __expf(s - m_i) : 0.0f;
          pa[g * 8 + w] = (__bf16)p;
        }
      }
      acc2[0] = WMMA_BF16(pa, vA0, acc2[0]);
      acc2[1] = WMMA_BF16(pa, vA1, acc2[1]);
      acc2[2] = WMMA_BF16(pa, vA2, acc2[2]);
      acc2[3] = WMMA_BF16(pa, vA3, acc2[3]);
    }
    if (j0 + 512 < 2048) {
      vA0 = load_b_frag(vb + (size_t)0 * 2048 + j0 + 512, 2048);
      vA1 = load_b_frag(vb + (size_t)16 * 2048 + j0 + 512, 2048);
      vA2 = load_b_frag(vb + (size_t)32 * 2048 + j0 + 512, 2048);
      vA3 = load_b_frag(vb + (size_t)48 * 2048 + j0 + 512, 2048);
    }
    {
      const int j0b = j0 + 256;
      v16bf pa;
#pragma unroll
      for (int g = 0; g < 2; ++g) {
#pragma unroll
        for (int w = 0; w < 8; ++w) {
          const int j = j0b + g * 16 + half * 8 + w;
          const float s = sc[cb * 2048 + j];
          const float p = (s >= th_i) ? __expf(s - m_i) : 0.0f;
          pa[g * 8 + w] = (__bf16)p;
        }
      }
      acc2[0] = WMMA_BF16(pa, vB0, acc2[0]);
      acc2[1] = WMMA_BF16(pa, vB1, acc2[1]);
      acc2[2] = WMMA_BF16(pa, vB2, acc2[2]);
      acc2[3] = WMMA_BF16(pa, vB3, acc2[3]);
    }
    if (j0 + 768 < 2048) {
      vB0 = load_b_frag(vb + (size_t)0 * 2048 + j0 + 768, 2048);
      vB1 = load_b_frag(vb + (size_t)16 * 2048 + j0 + 768, 2048);
      vB2 = load_b_frag(vb + (size_t)32 * 2048 + j0 + 768, 2048);
      vB3 = load_b_frag(vb + (size_t)48 * 2048 + j0 + 768, 2048);
    }
  }
#pragma unroll
  for (int nt = 0; nt < 4; ++nt) {
#pragma unroll
    for (int r = 0; r < 8; ++r)
      atomicAdd(&outacc[(rh + r) * 64 + nt * 16 + cb], acc2[nt][r]);
  }
  __syncthreads();

  const int b = bh >> 4, h = bh & 15;
  for (int idx = threadIdx.x; idx < 1024; idx += 256) {
    const int row = idx >> 6, col = idx & 63;
    const int n = i0 + row;
    const float v = outacc[idx] * rstat[32 + row];
    headout[((size_t)b * 2048 + n) * 1024 + h * 64 + col] = (__bf16)v;
  }
}

// ---------------------------------------------------------------------------
// Kernel 3: output projection.  out[t,c] = sum_e ho[t,e]*W_out[c,e] + b_out[c]
// M=4096, N=1024, K=1024. Same 32x64-per-wave ping-pong tiling.
// ---------------------------------------------------------------------------
__global__ __launch_bounds__(256) void out_gemm(const __bf16* __restrict__ ho,
                                                const __bf16* __restrict__ wo,
                                                const float* __restrict__ bias,
                                                float* __restrict__ out) {
  const int col0 = blockIdx.x * 64;
  const int wave = threadIdx.x >> 5;
  const int row0 = blockIdx.y * 256 + wave * 32;
  const __bf16* Abase = ho + (size_t)row0 * 1024;
  const __bf16* Bbase = wo + (size_t)col0 * 1024;

  v8f acc[8] = {};
  v16bf a0A = load_a_frag(Abase, 1024);
  v16bf a1A = load_a_frag(Abase + (size_t)16 * 1024, 1024);
  v16bf b0A = load_b_frag(Bbase, 1024);
  v16bf b1A = load_b_frag(Bbase + (size_t)16 * 1024, 1024);
  v16bf b2A = load_b_frag(Bbase + (size_t)32 * 1024, 1024);
  v16bf b3A = load_b_frag(Bbase + (size_t)48 * 1024, 1024);
  v16bf a0B = load_a_frag(Abase + 32, 1024);
  v16bf a1B = load_a_frag(Abase + (size_t)16 * 1024 + 32, 1024);
  v16bf b0B = load_b_frag(Bbase + 32, 1024);
  v16bf b1B = load_b_frag(Bbase + (size_t)16 * 1024 + 32, 1024);
  v16bf b2B = load_b_frag(Bbase + (size_t)32 * 1024 + 32, 1024);
  v16bf b3B = load_b_frag(Bbase + (size_t)48 * 1024 + 32, 1024);

  for (int k0 = 64; k0 < 1024; k0 += 64) {
    __builtin_prefetch(Bbase + k0 + 64, 0, 1);
    GEMM8(acc, a0A, a1A, b0A, b1A, b2A, b3A);
    a0A = load_a_frag(Abase + k0, 1024);
    a1A = load_a_frag(Abase + (size_t)16 * 1024 + k0, 1024);
    b0A = load_b_frag(Bbase + k0, 1024);
    b1A = load_b_frag(Bbase + (size_t)16 * 1024 + k0, 1024);
    b2A = load_b_frag(Bbase + (size_t)32 * 1024 + k0, 1024);
    b3A = load_b_frag(Bbase + (size_t)48 * 1024 + k0, 1024);
    GEMM8(acc, a0B, a1B, b0B, b1B, b2B, b3B);
    a0B = load_a_frag(Abase + k0 + 32, 1024);
    a1B = load_a_frag(Abase + (size_t)16 * 1024 + k0 + 32, 1024);
    b0B = load_b_frag(Bbase + k0 + 32, 1024);
    b1B = load_b_frag(Bbase + (size_t)16 * 1024 + k0 + 32, 1024);
    b2B = load_b_frag(Bbase + (size_t)32 * 1024 + k0 + 32, 1024);
    b3B = load_b_frag(Bbase + (size_t)48 * 1024 + k0 + 32, 1024);
  }
  GEMM8(acc, a0A, a1A, b0A, b1A, b2A, b3A);
  GEMM8(acc, a0B, a1B, b0B, b1B, b2B, b3B);

  const int lane = threadIdx.x & 31;
  const int cb = lane & 15;
  const int rh = (lane >> 4) * 8;
#pragma unroll
  for (int mt = 0; mt < 2; ++mt) {
#pragma unroll
    for (int nt = 0; nt < 4; ++nt) {
#pragma unroll
      for (int r = 0; r < 8; ++r) {
        const int t = row0 + mt * 16 + rh + r;
        const int e = col0 + nt * 16 + cb;
        out[(size_t)t * 1024 + e] = acc[mt * 4 + nt][r] + bias[e];
      }
    }
  }
}

// ---------------------------------------------------------------------------
// Host launcher
// ---------------------------------------------------------------------------
extern "C" void kernel_launch(void* const* d_in, const int* in_sizes, int n_in,
                              void* d_out, int out_size, void* d_ws, size_t ws_size,
                              hipStream_t stream) {
  const float* x    = (const float*)d_in[0];   // (2,2048,1024)
  const float* Wqkv = (const float*)d_in[1];   // (3072,1024)
  const float* Wout = (const float*)d_in[2];   // (1024,1024)
  const float* bout = (const float*)d_in[3];   // (1024,)
  float* out = (float*)d_out;

  char* ws = (char*)d_ws;
  __bf16* xb   = (__bf16*)(ws);                  //  8 MB  x bf16
  __bf16* wqb  = (__bf16*)(ws + 8388608);        //  6 MB  W_qkv bf16
  __bf16* wob  = (__bf16*)(ws + 14680064);       //  2 MB  W_out bf16
  __bf16* qbuf = (__bf16*)(ws + 16777216);       //  8 MB  q  [bh, n, 64]
  __bf16* kbuf = (__bf16*)(ws + 25165824);       //  8 MB  k  [bh, n, 64]
  __bf16* vT   = (__bf16*)(ws + 33554432);       //  8 MB  relu(v)^T [bh, 64, n]
  __bf16* ho   = (__bf16*)(ws + 41943040);       //  8 MB  head output [t, 1024]

  cvt_bf16<<<16384, 256, 0, stream>>>(x,    xb,  4194304);
  cvt_bf16<<<12288, 256, 0, stream>>>(Wqkv, wqb, 3145728);
  cvt_bf16<<<4096,  256, 0, stream>>>(Wout, wob, 1048576);

  qkv_gemm<<<dim3(48, 16), 256, 0, stream>>>(xb, wqb, qbuf, kbuf, vT);

  const size_t shmem = 16 * 2048 * 4 + 48 * 4 + 16 * 64 * 4; // 135360 B (< 320 KB WGP LDS)
  (void)hipFuncSetAttribute((const void*)attn_kernel,
                            hipFuncAttributeMaxDynamicSharedMemorySize, (int)shmem);
  attn_kernel<<<4096, 256, shmem, stream>>>(qbuf, kbuf, vT, ho);

  out_gemm<<<dim3(16, 16), 256, 0, stream>>>(ho, wob, bout, out);
}
